// get_model_20478404067985
// MI455X (gfx1250) — compile-verified, matrix operand order checked by reading, not run on hardware
//
#include <hip/hip_runtime.h>
#include <math.h>

// ---------------------------------------------------------------------------
// Model constants (B=1)
// ---------------------------------------------------------------------------
#define AA      5
#define A2      25
#define CC      64
#define SS      128
#define HW      576          // 24*24
#define NTOK    14400        // A2*HW
#define NLAYER  4
#define HHR     48           // H*SCALE
#define WHR     48
#define NPIXHR  2304         // 48*48
#define FEATHR  1600         // A2*CC
#define INDIM   1652         // FEATHR + 2 + 2*A2

#define CDIV(a, b) (((a) + (b) - 1) / (b))

typedef __attribute__((ext_vector_type(16))) _Float16 v16h;
typedef __attribute__((ext_vector_type(8)))  float    v8f;

// ---------------------------------------------------------------------------
// Generic WMMA GEMM:  C[M,N] = A[M,K] @ W[N,K]^T  (+bias) (ReLU) (+resid)
// One wave -> one 16x16 tile, K stepped by 32.
// Fast path (tile fully in-bounds, full K step): each lane loads its 16
// fragment elements as two contiguous 8-float runs -> 4x global_load_b128.
// Slow path (edge tiles / K tail): clamped addresses + value masks, fully
// branchless so no per-element exec save/restore.
// Fragment layouts per CDNA5 ISA 7.12.2 (16-bit A 16x32; f32 C/D 16x16).
// ---------------------------------------------------------------------------
__global__ __launch_bounds__(256) void wmma_gemm(
    const float* __restrict__ Am, const float* __restrict__ Wm,
    float* __restrict__ Cm, const float* __restrict__ bias,
    const float* __restrict__ resid, int M, int N, int K, int act)
{
    const int lane   = threadIdx.x & 31;
    const int wave   = blockIdx.x * (blockDim.x >> 5) + (threadIdx.x >> 5);
    const int tilesN = (N + 15) >> 4;
    const int tilesM = (M + 15) >> 4;
    if (wave >= tilesM * tilesN) return;
    const int tm   = (wave / tilesN) << 4;
    const int tn   = (wave % tilesN) << 4;
    const int half = lane >> 4;
    const int idx  = lane & 15;
    const int arow = tm + idx;      // A row this lane supplies
    const int bcol = tn + idx;      // B column (row of W) this lane supplies
    const bool fullTile = (tm + 16 <= M) && (tn + 16 <= N);

    v8f acc = {};
    for (int k0 = 0; k0 < K; k0 += 32) {
        v16h af, bf;
        if (fullTile && (k0 + 32 <= K)) {
            // -------- fast path: vectorized 16B loads, no guards ----------
            const float* ap = Am + (size_t)arow * K + k0 + (half << 3);
            const float* bp = Wm + (size_t)bcol * K + k0 + (half << 3);
            if (k0 + 64 <= K) {                 // prefetch next K step
                __builtin_prefetch(ap + 32, 0, 3);
                __builtin_prefetch(bp + 32, 0, 3);
            }
            float4 a0 = *(const float4*)(ap);
            float4 a1 = *(const float4*)(ap + 4);
            float4 a2 = *(const float4*)(ap + 16);
            float4 a3 = *(const float4*)(ap + 20);
            float4 b0 = *(const float4*)(bp);
            float4 b1 = *(const float4*)(bp + 4);
            float4 b2 = *(const float4*)(bp + 16);
            float4 b3 = *(const float4*)(bp + 20);
            af[0]  = (_Float16)a0.x; af[1]  = (_Float16)a0.y;
            af[2]  = (_Float16)a0.z; af[3]  = (_Float16)a0.w;
            af[4]  = (_Float16)a1.x; af[5]  = (_Float16)a1.y;
            af[6]  = (_Float16)a1.z; af[7]  = (_Float16)a1.w;
            af[8]  = (_Float16)a2.x; af[9]  = (_Float16)a2.y;
            af[10] = (_Float16)a2.z; af[11] = (_Float16)a2.w;
            af[12] = (_Float16)a3.x; af[13] = (_Float16)a3.y;
            af[14] = (_Float16)a3.z; af[15] = (_Float16)a3.w;
            bf[0]  = (_Float16)b0.x; bf[1]  = (_Float16)b0.y;
            bf[2]  = (_Float16)b0.z; bf[3]  = (_Float16)b0.w;
            bf[4]  = (_Float16)b1.x; bf[5]  = (_Float16)b1.y;
            bf[6]  = (_Float16)b1.z; bf[7]  = (_Float16)b1.w;
            bf[8]  = (_Float16)b2.x; bf[9]  = (_Float16)b2.y;
            bf[10] = (_Float16)b2.z; bf[11] = (_Float16)b2.w;
            bf[12] = (_Float16)b3.x; bf[13] = (_Float16)b3.y;
            bf[14] = (_Float16)b3.z; bf[15] = (_Float16)b3.w;
        } else {
            // -------- slow path: clamped address + value mask (branchless)
            const int ar = arow < M ? arow : M - 1;
            const int bc = bcol < N ? bcol : N - 1;
            const float maskA = arow < M ? 1.f : 0.f;
            const float maskB = bcol < N ? 1.f : 0.f;
#pragma unroll
            for (int v = 0; v < 8; ++v) {
                const int kk = k0 + ((v >> 2) << 4) + (half << 3) + ((v & 3) << 1);
                const int kc0 = kk < K ? kk : K - 1;
                const int kc1 = kk + 1 < K ? kk + 1 : K - 1;
                const float mk0 = kk < K ? 1.f : 0.f;
                const float mk1 = kk + 1 < K ? 1.f : 0.f;
                af[2 * v]     = (_Float16)(Am[(size_t)ar * K + kc0] * (maskA * mk0));
                af[2 * v + 1] = (_Float16)(Am[(size_t)ar * K + kc1] * (maskA * mk1));
                bf[2 * v]     = (_Float16)(Wm[(size_t)bc * K + kc0] * (maskB * mk0));
                bf[2 * v + 1] = (_Float16)(Wm[(size_t)bc * K + kc1] * (maskB * mk1));
            }
        }
        acc = __builtin_amdgcn_wmma_f32_16x16x32_f16(
                  false, af, false, bf, (short)0, acc, false, false);
    }
#pragma unroll
    for (int r = 0; r < 8; ++r) {
        const int m = tm + r + (half << 3);
        const int n = tn + idx;
        if (m < M && n < N) {
            float vv = acc[r];
            if (bias)  vv += bias[n];
            if (act)   vv = fmaxf(vv, 0.f);
            if (resid) vv += resid[(size_t)m * N + n];
            Cm[(size_t)m * N + n] = vv;
        }
    }
}

// ---------------------------------------------------------------------------
// WMMA GEMM with fused 3x3 im2col A-gather (replaces unfold3x3 @ spa_mlp.T).
// A[m,k]: m = pix*NV + view ; k = c*9 + kh*3 + kw ; src is [C][NV][24][24].
// All launches are tile-exact (M in {14400,576}, N=128, K=576) so there are
// no M/N/K guards; the zero-pad border test is a branchless clamp + mask.
// ---------------------------------------------------------------------------
__device__ __forceinline__ float im2col_a(const float* __restrict__ src,
                                          int NV, int m, int k)
{
    const int pix  = m / NV;
    const int view = m - pix * NV;
    const int c    = k / 9;
    const int rem  = k - c * 9;
    const int kh   = rem / 3;
    const int kw   = rem - kh * 3;
    const int pi   = pix / 24;
    const int pj   = pix - pi * 24;
    const int i    = pi + kh - 1;
    const int j    = pj + kw - 1;
    const int ic   = i < 0 ? 0 : (i > 23 ? 23 : i);
    const int jc   = j < 0 ? 0 : (j > 23 ? 23 : j);
    const float v  = src[(((size_t)c * NV + view) * 24 + ic) * 24 + jc];
    return (i == ic && j == jc) ? v : 0.f;
}

__global__ __launch_bounds__(256) void wmma_gemm_unfold(
    const float* __restrict__ src, const float* __restrict__ Wm,
    float* __restrict__ Cm, int NV, int M, int N, int K)
{
    const int lane   = threadIdx.x & 31;
    const int wave   = blockIdx.x * (blockDim.x >> 5) + (threadIdx.x >> 5);
    const int tilesN = N >> 4;
    const int tilesM = M >> 4;
    if (wave >= tilesM * tilesN) return;
    const int tm   = (wave / tilesN) << 4;
    const int tn   = (wave % tilesN) << 4;
    const int half = lane >> 4;
    const int idx  = lane & 15;
    const int arow = tm + idx;
    const int bcol = tn + idx;

    v8f acc = {};
    for (int k0 = 0; k0 < K; k0 += 32) {
        v16h af, bf;
        // B side: contiguous weight rows -> vectorized 16B loads
        const float* bp = Wm + (size_t)bcol * K + k0 + (half << 3);
        float4 b0 = *(const float4*)(bp);
        float4 b1 = *(const float4*)(bp + 4);
        float4 b2 = *(const float4*)(bp + 16);
        float4 b3 = *(const float4*)(bp + 20);
        bf[0]  = (_Float16)b0.x; bf[1]  = (_Float16)b0.y;
        bf[2]  = (_Float16)b0.z; bf[3]  = (_Float16)b0.w;
        bf[4]  = (_Float16)b1.x; bf[5]  = (_Float16)b1.y;
        bf[6]  = (_Float16)b1.z; bf[7]  = (_Float16)b1.w;
        bf[8]  = (_Float16)b2.x; bf[9]  = (_Float16)b2.y;
        bf[10] = (_Float16)b2.z; bf[11] = (_Float16)b2.w;
        bf[12] = (_Float16)b3.x; bf[13] = (_Float16)b3.y;
        bf[14] = (_Float16)b3.z; bf[15] = (_Float16)b3.w;
        // A side: im2col gather (branchless pad handling)
#pragma unroll
        for (int v = 0; v < 8; ++v) {
            const int kk = k0 + ((v >> 2) << 4) + (half << 3) + ((v & 3) << 1);
            af[2 * v]     = (_Float16)im2col_a(src, NV, arow, kk);
            af[2 * v + 1] = (_Float16)im2col_a(src, NV, arow, kk + 1);
        }
        acc = __builtin_amdgcn_wmma_f32_16x16x32_f16(
                  false, af, false, bf, (short)0, acc, false, false);
    }
#pragma unroll
    for (int r = 0; r < 8; ++r) {
        const int m = tm + r + (half << 3);
        const int n = tn + idx;
        Cm[(size_t)m * N + n] = acc[r];
    }
}

// ---------------------------------------------------------------------------
// Positional encodings
// ---------------------------------------------------------------------------
__device__ __forceinline__ float pe1d(float pos, int c, int d)
{
    const int half = d >> 1;
    if (c < half) {
        float g = __powf(10000.f, 2.f * (float)c / (float)d);
        return __sinf(pos / g);
    } else {
        int j = c - half;
        float g = __powf(10000.f, 2.f * (float)j / (float)d);
        return __cosf(pos / g);
    }
}

__global__ void k_pe_ang(float* __restrict__ out)   // [A2, CC]
{
    int t = blockIdx.x * blockDim.x + threadIdx.x;
    if (t >= A2 * CC) return;
    int a = t / CC, c = t - a * CC;
    out[t] = pe1d((float)a, c, CC);
}

__global__ void k_pe_spa(float* __restrict__ out)   // [CC][24][24]
{
    int t = blockIdx.x * blockDim.x + threadIdx.x;
    if (t >= CC * HW) return;
    int c = t / HW, p = t - c * HW;
    int i = p / 24, j = p - i * 24;
    out[t] = 0.5f * (pe1d((float)i, c, CC) + pe1d((float)j, c, CC));
}

// ---------------------------------------------------------------------------
// Conv stem.  buf layout: [CC][A2][24][24].
// ---------------------------------------------------------------------------
__global__ void k_conv0(const float* __restrict__ lr,
                        const float* __restrict__ w0,
                        float* __restrict__ dst)
{
    int t = blockIdx.x * blockDim.x + threadIdx.x;
    if (t >= CC * A2 * HW) return;
    int c = t / NTOK, r = t - c * NTOK;
    int a = r / HW, p = r - a * HW;
    int i = p / 24, j = p - i * 24;
    int a1 = a / AA, a2 = a - a1 * AA;
    float acc = 0.f;
#pragma unroll
    for (int kh = 0; kh < 3; ++kh)
#pragma unroll
        for (int kw = 0; kw < 3; ++kw) {
            int ii = i + kh - 1, jj = j + kw - 1;
            if (ii < 0 || ii >= 24 || jj < 0 || jj >= 24) continue;
            acc += lr[(a1 * 24 + ii) * 120 + a2 * 24 + jj] * w0[c * 9 + kh * 3 + kw];
        }
    dst[t] = acc;
}

__global__ void k_convi(const float* __restrict__ src,
                        const float* __restrict__ wt,
                        float* __restrict__ dst)
{
    int t = blockIdx.x * blockDim.x + threadIdx.x;
    if (t >= CC * A2 * HW) return;
    int co = t / NTOK, r = t - co * NTOK;
    int a = r / HW, p = r - a * HW;
    int i = p / 24, j = p - i * 24;
    float acc = 0.f;
    for (int ci = 0; ci < CC; ++ci) {
        const float* sv = src + (size_t)ci * NTOK + (size_t)a * HW;
        const float* wv = wt + ((size_t)co * CC + ci) * 9;
#pragma unroll
        for (int kh = 0; kh < 3; ++kh)
#pragma unroll
            for (int kw = 0; kw < 3; ++kw) {
                int ii = i + kh - 1, jj = j + kw - 1;
                if (ii < 0 || ii >= 24 || jj < 0 || jj >= 24) continue;
                acc += sv[ii * 24 + jj] * wv[kh * 3 + kw];
            }
    }
    dst[t] = acc > 0.f ? acc : 0.2f * acc;   // leaky_relu(0.2)
}

__global__ void k_add_inplace(float* __restrict__ dst,
                              const float* __restrict__ src, int n)
{
    int t = blockIdx.x * blockDim.x + threadIdx.x;
    if (t < n) dst[t] += src[t];
}

// ---------------------------------------------------------------------------
// Token layout transposes (ang tokens: rows m = a*576+p, D = CC)
// ---------------------------------------------------------------------------
__global__ void k_blk_to_tok(const float* __restrict__ blk, float* __restrict__ tok)
{
    int t = blockIdx.x * blockDim.x + threadIdx.x;
    if (t >= NTOK * CC) return;
    int row = t / CC, c = t - row * CC;
    int a = row / HW, p = row - a * HW;
    tok[t] = blk[(size_t)c * NTOK + (size_t)a * HW + p];
}

__global__ void k_tok_to_blk(const float* __restrict__ tok, float* __restrict__ blk)
{
    int t = blockIdx.x * blockDim.x + threadIdx.x;
    if (t >= NTOK * CC) return;
    int c = t / NTOK, r = t - c * NTOK;
    int a = r / HW, p = r - a * HW;
    blk[t] = tok[((size_t)a * HW + p) * CC + c];
}

// ---------------------------------------------------------------------------
// LayerNorm over last dim D with optional additive PE (pe row = row / peDiv)
// ---------------------------------------------------------------------------
__global__ void k_layernorm(const float* __restrict__ x, const float* __restrict__ pe,
                            int peDiv, const float* __restrict__ g,
                            const float* __restrict__ b, float* __restrict__ y,
                            int rows, int D)
{
    int r = blockIdx.x * blockDim.x + threadIdx.x;
    if (r >= rows) return;
    const float* xr = x + (size_t)r * D;
    const float* pr = pe ? pe + (size_t)(r / peDiv) * D : nullptr;
    float mean = 0.f;
    for (int i = 0; i < D; ++i) mean += xr[i] + (pr ? pr[i] : 0.f);
    mean /= (float)D;
    float var = 0.f;
    for (int i = 0; i < D; ++i) {
        float d = xr[i] + (pr ? pr[i] : 0.f) - mean;
        var += d * d;
    }
    var /= (float)D;
    float inv = rsqrtf(var + 1e-5f);
    float* yr = y + (size_t)r * D;
    for (int i = 0; i < D; ++i)
        yr[i] = (xr[i] + (pr ? pr[i] : 0.f) - mean) * inv * g[i] + b[i];
}

// ---------------------------------------------------------------------------
// Angular attention: tokens [25, 576, 64], nh=8, dh=8.
// qk rows [q(64)|k(64)], one thread per (a, p, h), 2-pass softmax.
// ---------------------------------------------------------------------------
__global__ void k_ang_attention(const float* __restrict__ qk,
                                const float* __restrict__ vb,
                                float* __restrict__ att)
{
    int t = blockIdx.x * blockDim.x + threadIdx.x;
    if (t >= A2 * HW * 8) return;
    int a = t / (HW * 8), r = t - a * (HW * 8);
    int p = r / 8, h = r - p * 8;
    const float* qrow = qk + ((size_t)a * HW + p) * 128 + h * 8;
    float q[8];
#pragma unroll
    for (int d = 0; d < 8; ++d) q[d] = qrow[d];
    float mx = -1e30f;
    for (int m = 0; m < A2; ++m) {
        const float* kr = qk + ((size_t)m * HW + p) * 128 + 64 + h * 8;
        float s = 0.f;
#pragma unroll
        for (int d = 0; d < 8; ++d) s += q[d] * kr[d];
        s *= 0.35355339f;                       // 1/sqrt(8)
        mx = fmaxf(mx, s);
    }
    float Z = 0.f, o[8] = {};
    for (int m = 0; m < A2; ++m) {
        const float* kr = qk + ((size_t)m * HW + p) * 128 + 64 + h * 8;
        float s = 0.f;
#pragma unroll
        for (int d = 0; d < 8; ++d) s += q[d] * kr[d];
        float e = __expf(s * 0.35355339f - mx);
        Z += e;
        const float* vr = vb + ((size_t)m * HW + p) * 64 + h * 8;
#pragma unroll
        for (int d = 0; d < 8; ++d) o[d] += e * vr[d];
    }
    float inv = 1.f / Z;
    float* orow = att + ((size_t)a * HW + p) * 64 + h * 8;
#pragma unroll
    for (int d = 0; d < 8; ++d) orow[d] = o[d] * inv;
}

// ---------------------------------------------------------------------------
// Spatial windowed attention: tokens [576(pix), 25(view), 128], nh=8, dh=16.
// window_mask(24,24,5): keys with |di|<=2, |dj|<=2 (clamped) only.
// ---------------------------------------------------------------------------
__global__ void k_spa_attention(const float* __restrict__ qk,
                                const float* __restrict__ vb,
                                float* __restrict__ att)
{
    int t = blockIdx.x * blockDim.x + threadIdx.x;
    if (t >= A2 * HW * 8) return;
    int a = t / (HW * 8), r = t - a * (HW * 8);
    int pix = r / 8, h = r - pix * 8;
    int i = pix / 24, j = pix - i * 24;
    int i0 = i - 2 < 0 ? 0 : i - 2, i1 = i + 2 > 23 ? 23 : i + 2;
    int j0 = j - 2 < 0 ? 0 : j - 2, j1 = j + 2 > 23 ? 23 : j + 2;
    const float* qrow = qk + ((size_t)pix * A2 + a) * 256 + h * 16;
    float q[16];
#pragma unroll
    for (int d = 0; d < 16; ++d) q[d] = qrow[d];
    float mx = -1e30f;
    for (int i2 = i0; i2 <= i1; ++i2)
        for (int j2 = j0; j2 <= j1; ++j2) {
            const float* kr = qk + (((size_t)(i2 * 24 + j2)) * A2 + a) * 256 + 128 + h * 16;
            float s = 0.f;
#pragma unroll
            for (int d = 0; d < 16; ++d) s += q[d] * kr[d];
            mx = fmaxf(mx, s * 0.25f);          // 1/sqrt(16)
        }
    float Z = 0.f, o[16] = {};
    for (int i2 = i0; i2 <= i1; ++i2)
        for (int j2 = j0; j2 <= j1; ++j2) {
            int p2 = i2 * 24 + j2;
            const float* kr = qk + (((size_t)p2) * A2 + a) * 256 + 128 + h * 16;
            float s = 0.f;
#pragma unroll
            for (int d = 0; d < 16; ++d) s += q[d] * kr[d];
            float e = __expf(s * 0.25f - mx);
            Z += e;
            const float* vr = vb + (((size_t)p2) * A2 + a) * 128 + h * 16;
#pragma unroll
            for (int d = 0; d < 16; ++d) o[d] += e * vr[d];
        }
    float inv = 1.f / Z;
    float* orow = att + ((size_t)pix * A2 + a) * 128 + h * 16;
#pragma unroll
    for (int d = 0; d < 16; ++d) orow[d] = o[d] * inv;
}

// ---------------------------------------------------------------------------
// spa_lin 1x1x1 conv:  blk[o, a, pix] = sum_s lin[o,s] * tok[pix*25+a, s]
// ---------------------------------------------------------------------------
__global__ void k_spa_project(const float* __restrict__ lin,
                              const float* __restrict__ tok,
                              float* __restrict__ blk)
{
    int t = blockIdx.x * blockDim.x + threadIdx.x;
    if (t >= CC * A2 * HW) return;
    int o = t / NTOK, r = t - o * NTOK;
    int a = r / HW, pix = r - a * HW;
    const float* lr = lin + (size_t)o * SS;
    const float* tr = tok + ((size_t)pix * A2 + a) * SS;
    float acc = 0.f;
    for (int s = 0; s < SS; ++s) acc += lr[s] * tr[s];
    blk[t] = acc;
}

// ---------------------------------------------------------------------------
// Per-view bilinear 2x upsample (half-pixel, edge clamp) written straight into
// the imp-MLP input at column a*64+c.
// ---------------------------------------------------------------------------
__global__ void k_upsample_inp(const float* __restrict__ buf, float* __restrict__ inp)
{
    int t = blockIdx.x * blockDim.x + threadIdx.x;
    if (t >= NPIXHR * FEATHR) return;
    int pix = t / FEATHR, k = t - pix * FEATHR;
    int a = k / CC, c = k - a * CC;
    int y = pix / WHR, x = pix - y * WHR;
    float sy = (y + 0.5f) * 0.5f - 0.5f;
    float sx = (x + 0.5f) * 0.5f - 0.5f;
    int y0 = (int)floorf(sy), x0 = (int)floorf(sx);
    float fy = sy - y0, fx = sx - x0;
    int ya = y0 < 0 ? 0 : (y0 > 23 ? 23 : y0);
    int yb = y0 + 1 < 0 ? 0 : (y0 + 1 > 23 ? 23 : y0 + 1);
    int xa = x0 < 0 ? 0 : (x0 > 23 ? 23 : x0);
    int xb = x0 + 1 < 0 ? 0 : (x0 + 1 > 23 ? 23 : x0 + 1);
    const float* s = buf + ((size_t)c * A2 + a) * HW;
    float v = (1.f - fy) * ((1.f - fx) * s[ya * 24 + xa] + fx * s[ya * 24 + xb]) +
              fy * ((1.f - fx) * s[yb * 24 + xa] + fx * s[yb * 24 + xb]);
    inp[(size_t)pix * INDIM + a * CC + c] = v;
}

// ---------------------------------------------------------------------------
// Fill imp input columns 1600..1651: rel coords (from nearest-grid LR sample)
// then angular-cell constants.
// ---------------------------------------------------------------------------
__global__ void k_inp_tail(const float* __restrict__ coord, float* __restrict__ inp)
{
    int t = blockIdx.x * blockDim.x + threadIdx.x;
    if (t >= NPIXHR * 52) return;
    int p = t / 52, k = t - p * 52;
    float val;
    if (k < 2) {
        float c0 = coord[(size_t)p * 2 + k];
        float cc = fminf(fmaxf(c0 + 1e-6f, -1.f + 1e-6f), 1.f - 1e-6f);
        float ih = roundf(((cc + 1.f) * 24.f - 1.f) * 0.5f);
        ih = fminf(fmaxf(ih, 0.f), 23.f);
        float qv = -1.f + (2.f * ih + 1.f) / 24.f;
        val = (c0 - qv) * 24.f;
    } else {
        int kk = k - 2, pi = kk >> 1, comp = kk & 1;
        int a1 = pi / AA, a2 = pi - a1 * AA;
        val = (comp == 0 ? (float)a1 : (float)a2) + 0.5f;
    }
    inp[(size_t)p * INDIM + 1600 + k] = val;
}

// ---------------------------------------------------------------------------
// Final rearrange: o[y,x,(a1,a2,ch)] -> out[ch, a1*48+y, a2*48+x]
// ---------------------------------------------------------------------------
__global__ void k_final(const float* __restrict__ o, float* __restrict__ out)
{
    int t = blockIdx.x * blockDim.x + threadIdx.x;
    if (t >= 3 * 240 * 240) return;
    int ch = t / 57600, rem = t - ch * 57600;
    int rr = rem / 240, cc = rem - rr * 240;
    int a1 = rr / HHR, y = rr - a1 * HHR;
    int a2 = cc / WHR, x = cc - a2 * WHR;
    out[t] = o[(((size_t)(y * WHR + x)) * A2 + a1 * AA + a2) * 3 + ch];
}

// ---------------------------------------------------------------------------
// Host orchestration
// ---------------------------------------------------------------------------
static void gemm(const float* A, const float* W, float* C, const float* bias,
                 const float* resid, int M, int N, int K, int act, hipStream_t s)
{
    int tiles = CDIV(M, 16) * CDIV(N, 16);
    wmma_gemm<<<CDIV(tiles, 8), 256, 0, s>>>(A, W, C, bias, resid, M, N, K, act);
}

static void gemm_unfold(const float* src, const float* W, float* C,
                        int NV, int M, int N, int K, hipStream_t s)
{
    int tiles = (M / 16) * (N / 16);
    wmma_gemm_unfold<<<CDIV(tiles, 8), 256, 0, s>>>(src, W, C, NV, M, N, K);
}

extern "C" void kernel_launch(void* const* d_in, const int* in_sizes, int n_in,
                              void* d_out, int out_size, void* d_ws, size_t ws_size,
                              hipStream_t stream)
{
    const float* lr       = (const float*)d_in[0];
    const float* coord    = (const float*)d_in[1];
    const float* conv0_w  = (const float*)d_in[2];
    const float* convi_w  = (const float*)d_in[3];
    const float* ang_ln_g = (const float*)d_in[4];
    const float* ang_ln_b = (const float*)d_in[5];
    const float* ang_in   = (const float*)d_in[6];
    const float* ang_out_w= (const float*)d_in[7];
    const float* ang_ff_g = (const float*)d_in[8];
    const float* ang_ff_b = (const float*)d_in[9];
    const float* ang_w1   = (const float*)d_in[10];
    const float* ang_w2   = (const float*)d_in[11];
    const float* spa_mlp  = (const float*)d_in[12];
    const float* spa_ln_g = (const float*)d_in[13];
    const float* spa_ln_b = (const float*)d_in[14];
    const float* spa_in   = (const float*)d_in[15];
    const float* spa_out_w= (const float*)d_in[16];
    const float* spa_ff_g = (const float*)d_in[17];
    const float* spa_ff_b = (const float*)d_in[18];
    const float* spa_w1   = (const float*)d_in[19];
    const float* spa_w2   = (const float*)d_in[20];
    const float* spa_linw = (const float*)d_in[21];
    const float* imp_w0   = (const float*)d_in[22];
    const float* imp_b0   = (const float*)d_in[23];
    const float* imp_w1   = (const float*)d_in[24];
    const float* imp_b1   = (const float*)d_in[25];
    const float* imp_w2   = (const float*)d_in[26];
    const float* imp_b2   = (const float*)d_in[27];
    float* out = (float*)d_out;
    float* ws  = (float*)d_ws;

    // ----- workspace layout (floats) -------------------------------------
    const size_t FEATSZ = (size_t)CC * A2 * HW;         // 921600
    float* buf    = ws;                                  // persistent feature
    float* blk    = buf    + FEATSZ;
    float* angpe  = blk    + FEATSZ;                     // 1600
    float* speimg = angpe  + 1600;                       // 36864
    float* petok  = speimg + (size_t)CC * HW;            // 73728
    float* tok    = petok  + (size_t)HW * SS;            // transient base
    float* tnb    = tok    + (size_t)NTOK * SS;
    float* qkb    = tnb    + (size_t)NTOK * SS;          // [NTOK, 2*SS] max
    float* vbuf   = qkb    + (size_t)NTOK * 2 * SS;
    float* attb   = vbuf   + (size_t)NTOK * SS;
    float* ffb    = attb   + (size_t)NTOK * SS;          // [NTOK, 2*SS] max
    // imp-phase overlay (transformer transients are dead by then)
    float* inp    = tok;                                 // [2304, 1652]
    float* h0     = tok + (size_t)NPIXHR * INDIM;
    float* h1     = h0  + (size_t)NPIXHR * 320;
    float* ob     = h1  + (size_t)NPIXHR * 320;          // [2304, 75]
    // conv ping-pong overlays tok/tn
    float* t0 = tok;
    float* t1 = tnb;

    const int TPB = 256;

    // ----- positional encodings ------------------------------------------
    k_pe_ang<<<CDIV(A2 * CC, TPB), TPB, 0, stream>>>(angpe);
    k_pe_spa<<<CDIV(CC * HW, TPB), TPB, 0, stream>>>(speimg);

    // ----- conv stem ------------------------------------------------------
    const int NFEAT = CC * A2 * HW;
    k_conv0<<<CDIV(NFEAT, TPB), TPB, 0, stream>>>(lr, conv0_w, buf);
    k_convi<<<CDIV(NFEAT, TPB), TPB, 0, stream>>>(buf, convi_w + 0 * CC * CC * 9, t0);
    k_convi<<<CDIV(NFEAT, TPB), TPB, 0, stream>>>(t0,  convi_w + 1 * CC * CC * 9, t1);
    k_convi<<<CDIV(NFEAT, TPB), TPB, 0, stream>>>(t1,  convi_w + 2 * CC * CC * 9, t0);
    k_add_inplace<<<CDIV(NFEAT, TPB), TPB, 0, stream>>>(buf, t0, NFEAT);
    (void)hipMemcpyAsync(blk, buf, FEATSZ * sizeof(float),
                         hipMemcpyDeviceToDevice, stream);

    // ----- transformer layers --------------------------------------------
    for (int l = 0; l < NLAYER; ++l) {
        // ============ AngTrans: tokens [25, 576, 64] ============
        k_blk_to_tok<<<CDIV(NFEAT, TPB), TPB, 0, stream>>>(blk, tok);
        k_layernorm<<<CDIV(NTOK, TPB), TPB, 0, stream>>>(
            tok, angpe, HW, ang_ln_g + l * CC, ang_ln_b + l * CC, tnb, NTOK, CC);
        // q|k from tn, v from tok
        gemm(tnb, ang_in + (size_t)l * 3 * CC * CC, qkb, nullptr, nullptr,
             NTOK, 2 * CC, CC, 0, stream);
        gemm(tok, ang_in + (size_t)l * 3 * CC * CC + 2 * CC * CC, vbuf, nullptr, nullptr,
             NTOK, CC, CC, 0, stream);
        k_ang_attention<<<CDIV(A2 * HW * 8, TPB), TPB, 0, stream>>>(qkb, vbuf, attb);
        gemm(attb, ang_out_w + (size_t)l * CC * CC, tok, nullptr, tok,
             NTOK, CC, CC, 0, stream);                         // +residual
        k_layernorm<<<CDIV(NTOK, TPB), TPB, 0, stream>>>(
            tok, nullptr, 1, ang_ff_g + l * CC, ang_ff_b + l * CC, tnb, NTOK, CC);
        gemm(tnb, ang_w1 + (size_t)l * 2 * CC * CC, ffb, nullptr, nullptr,
             NTOK, 2 * CC, CC, 1, stream);                     // ReLU
        gemm(ffb, ang_w2 + (size_t)l * 2 * CC * CC, tok, nullptr, tok,
             NTOK, CC, 2 * CC, 0, stream);                     // +residual
        k_tok_to_blk<<<CDIV(NFEAT, TPB), TPB, 0, stream>>>(tok, blk);

        // ============ SpaTrans: tokens [576, 25, 128], 5x5 window ============
        gemm_unfold(blk, spa_mlp + (size_t)l * SS * 9 * CC, tok,
                    A2, NTOK, SS, 9 * CC, stream);             // tok = unfold@mlp^T
        gemm_unfold(speimg, spa_mlp + (size_t)l * SS * 9 * CC, petok,
                    1, HW, SS, 9 * CC, stream);                // pe tokens
        k_layernorm<<<CDIV(NTOK, TPB), TPB, 0, stream>>>(
            tok, petok, A2, spa_ln_g + l * SS, spa_ln_b + l * SS, tnb, NTOK, SS);
        gemm(tnb, spa_in + (size_t)l * 3 * SS * SS, qkb, nullptr, nullptr,
             NTOK, 2 * SS, SS, 0, stream);
        gemm(tok, spa_in + (size_t)l * 3 * SS * SS + 2 * SS * SS, vbuf, nullptr, nullptr,
             NTOK, SS, SS, 0, stream);
        k_spa_attention<<<CDIV(A2 * HW * 8, TPB), TPB, 0, stream>>>(qkb, vbuf, attb);
        gemm(attb, spa_out_w + (size_t)l * SS * SS, tok, nullptr, tok,
             NTOK, SS, SS, 0, stream);                         // +residual
        k_layernorm<<<CDIV(NTOK, TPB), TPB, 0, stream>>>(
            tok, nullptr, 1, spa_ff_g + l * SS, spa_ff_b + l * SS, tnb, NTOK, SS);
        gemm(tnb, spa_w1 + (size_t)l * 2 * SS * SS, ffb, nullptr, nullptr,
             NTOK, 2 * SS, SS, 1, stream);                     // ReLU
        gemm(ffb, spa_w2 + (size_t)l * 2 * SS * SS, tok, nullptr, tok,
             NTOK, SS, 2 * SS, 0, stream);                     // +residual
        k_spa_project<<<CDIV(NFEAT, TPB), TPB, 0, stream>>>(
            spa_linw + (size_t)l * CC * SS, tok, blk);
    }

    // ----- residual + implicit MLP decoder -------------------------------
    k_add_inplace<<<CDIV(NFEAT, TPB), TPB, 0, stream>>>(buf, blk, NFEAT);
    k_upsample_inp<<<CDIV(NPIXHR * FEATHR, TPB), TPB, 0, stream>>>(buf, inp);
    k_inp_tail<<<CDIV(NPIXHR * 52, TPB), TPB, 0, stream>>>(coord, inp);
    gemm(inp, imp_w0, h0, imp_b0, nullptr, NPIXHR, 320, INDIM, 1, stream);
    gemm(h0,  imp_w1, h1, imp_b1, nullptr, NPIXHR, 320, 320,   1, stream);
    gemm(h1,  imp_w2, ob, imp_b2, nullptr, NPIXHR, 75,  320,   0, stream);
    k_final<<<CDIV(3 * 240 * 240, TPB), TPB, 0, stream>>>(ob, out);

    (void)in_sizes; (void)n_in; (void)out_size; (void)ws_size;
}
